// ExpansionContrastModule_26697516712213
// MI455X (gfx1250) — matrix operand order repaired and checked by reading, not compiled, not run on hardware
//
#include <hip/hip_runtime.h>
#include <hip/hip_bf16.h>
#include <cstdint>

// ---------------- problem constants ----------------
#define BB_   4      // batch
#define CIN_  64
#define CC_   16     // C
#define HH_   256
#define WW_   256
#define EXP_  16
#define PIX_  (HH_*WW_)          // 65536
#define BN_EPS_ 1e-5f

typedef __attribute__((ext_vector_type(16))) _Float16 v16h;
typedef __attribute__((ext_vector_type(8)))  float    v8f;

// ---------------- TDM availability (device pass only) ----------------
#if defined(__HIP_DEVICE_COMPILE__) && __has_builtin(__builtin_amdgcn_tensor_load_to_lds)
#define HAVE_TDM 1
#else
#define HAVE_TDM 0
#endif

#if defined(__HIP_DEVICE_COMPILE__) && __has_builtin(__builtin_amdgcn_s_wait_tensorcnt)
#define WAIT_TENSORCNT0() __builtin_amdgcn_s_wait_tensorcnt(0)
#else
#define WAIT_TENSORCNT0() ((void)0)
#endif

#if HAVE_TDM
typedef __attribute__((ext_vector_type(4))) unsigned int u32x4_;
typedef __attribute__((ext_vector_type(8))) int          i32x8_;
typedef __attribute__((ext_vector_type(4))) int          i32x4_;
#endif

// Stage a contiguous 1-D float array (nelem <= 65535) from global into LDS.
// Uses the Tensor Data Mover (one DMA per workgroup, issued by wave 0) when
// available; falls back to a cooperative copy loop otherwise.
__device__ __forceinline__ void stage_to_lds(float* lds, const float* gsrc, int nelem) {
#if HAVE_TDM
  if (threadIdx.x < 32) {                       // wave-uniform: only wave 0 issues
    unsigned long long ga = (unsigned long long)(uintptr_t)gsrc;
    unsigned ldsoff = (unsigned)(uintptr_t)lds; // flat addr low 32 bits == LDS offset
    unsigned n = (unsigned)nelem;
    u32x4_ g0;
    g0[0] = 1u;                                  // count=1 (valid), user mode, no gather
    g0[1] = ldsoff;                              // lds_addr (bytes)
    g0[2] = (unsigned)(ga & 0xffffffffull);      // global_addr[31:0]
    g0[3] = (unsigned)((ga >> 32) & 0x01ffffffull) | (2u << 30); // ga[56:32] | type=2
    i32x8_ g1;
    g1[0] = (int)(2u << 16);                     // workgroup_mask=0, data_size=2 (4B)
    g1[1] = (int)((n & 0xffffu) << 16);          // tensor_dim0[15:0] in [63:48]
    g1[2] = (int)(((n >> 16) & 0xffffu) | (1u << 16)); // tensor_dim0[31:16], tensor_dim1=1
    g1[3] = (int)((n & 0xffffu) << 16);          // tensor_dim1 hi=0, tile_dim0=n
    g1[4] = 0;                                   // tile_dim1=0, tile_dim2=0 (unused)
    g1[5] = (int)n;                              // tensor_dim0_stride[31:0]
    g1[6] = (int)((n & 0xffffu) << 16);          // stride0 hi=0, tensor_dim1_stride lo16
    g1[7] = 0;
    i32x4_ gz = {0, 0, 0, 0};
#if __clang_major__ >= 23
    i32x8_ gz8 = {0, 0, 0, 0, 0, 0, 0, 0};
    __builtin_amdgcn_tensor_load_to_lds(g0, g1, gz, gz, gz8, 0);
#else
    __builtin_amdgcn_tensor_load_to_lds(g0, g1, gz, gz, 0);
#endif
    WAIT_TENSORCNT0();                           // TENSORcnt -> 0 before sharing
  }
  __syncthreads();
#else
  for (int i = (int)threadIdx.x; i < nelem; i += (int)blockDim.x) lds[i] = gsrc[i];
  __syncthreads();
#endif
}

// ---- WMMA 16-bit A fragment index: lane l, vector elem i (VGPR i/2, half i&1)
__device__ __forceinline__ int a_kidx(int lane, int i) {
  int v = i >> 1, hx = i & 1;
  int k = (v < 4) ? (2 * v + hx) : (16 + 2 * (v - 4) + hx);
  return k + ((lane < 16) ? 0 : 8);
}
// ---- WMMA 16-bit B fragment index (32xK chunk x 16 cols)
__device__ __forceinline__ int b_kidx(int lane, int i) {
  return ((lane < 16) ? 0 : 8) + ((i < 8) ? i : (8 + i));
}

__device__ __forceinline__ float sigmoidf_(float v) { return 1.f / (1.f + __expf(-v)); }

// ================= kernel 0: on-device softmaxes =================
__global__ void softmax_small_kernel(const float* __restrict__ scales1,
                                     const float* __restrict__ scales2,
                                     const float* __restrict__ scales3,
                                     float* __restrict__ s1,
                                     float* __restrict__ k3s,
                                     float* __restrict__ k4s) {
  int t = threadIdx.x;
  { // k3: [C*EXP=256][8] softmax over 8
    float v[8]; float mx = -1e30f;
    #pragma unroll
    for (int j = 0; j < 8; ++j) { v[j] = scales2[t * 8 + j]; mx = fmaxf(mx, v[j]); }
    float s = 0.f;
    #pragma unroll
    for (int j = 0; j < 8; ++j) { v[j] = __expf(v[j] - mx); s += v[j]; }
    float inv = 1.f / s;
    #pragma unroll
    for (int j = 0; j < 8; ++j) k3s[t * 8 + j] = v[j] * inv;
  }
  if (t < CC_) { // k4: [16][16] softmax over 16
    float v[16]; float mx = -1e30f;
    #pragma unroll
    for (int j = 0; j < 16; ++j) { v[j] = scales3[t * 16 + j]; mx = fmaxf(mx, v[j]); }
    float s = 0.f;
    #pragma unroll
    for (int j = 0; j < 16; ++j) { v[j] = __expf(v[j] - mx); s += v[j]; }
    float inv = 1.f / s;
    #pragma unroll
    for (int j = 0; j < 16; ++j) k4s[t * 16 + j] = v[j] * inv;
  }
  if (t == 0) { // scales1: softmax over 4
    float v[4]; float mx = -1e30f;
    #pragma unroll
    for (int j = 0; j < 4; ++j) { v[j] = scales1[j]; mx = fmaxf(mx, v[j]); }
    float s = 0.f;
    #pragma unroll
    for (int j = 0; j < 4; ++j) { v[j] = __expf(v[j] - mx); s += v[j]; }
    float inv = 1.f / s;
    #pragma unroll
    for (int j = 0; j < 4; ++j) s1[j] = v[j] * inv;
  }
}

// ================= kernel 1: 1x1 conv 64->16 via WMMA (pure GEMM) =================
// Weights (16x64 = 4KB) staged once per block via TDM; A-fragments from LDS.
__global__ void in_conv_wmma(const float* __restrict__ cen,
                             const float* __restrict__ w_in,
                             const float* __restrict__ b_in,
                             float* __restrict__ x) {
  __shared__ float wlds[CC_ * CIN_];
  stage_to_lds(wlds, w_in, CC_ * CIN_);

  int wave = (blockIdx.x * blockDim.x + threadIdx.x) >> 5;
  int lane = threadIdx.x & 31;
  const int tiles_per_b = PIX_ / 16;          // 4096
  int b  = wave / tiles_per_b;
  int p0 = (wave % tiles_per_b) * 16;
  int n  = lane & 15;

  // prefetch next tile's first input row (global_prefetch_b8)
  __builtin_prefetch(&cen[(size_t)b * CIN_ * PIX_ + p0 + 16], 0, 0);

  v8f acc = {};
  #pragma unroll
  for (int kc = 0; kc < CIN_; kc += 32) {
    v16h a, bf;
    #pragma unroll
    for (int i = 0; i < 16; ++i) {
      int ka = kc + a_kidx(lane, i);
      a[i]  = (_Float16)wlds[n * CIN_ + ka];
      int kb = kc + b_kidx(lane, i);
      bf[i] = (_Float16)cen[((size_t)b * CIN_ + kb) * PIX_ + p0 + n];
    }
    acc = __builtin_amdgcn_wmma_f32_16x16x32_f16(false, a, false, bf,
                                                 (short)0, acc, false, false);
  }
  int mb = (lane < 16) ? 0 : 8;
  #pragma unroll
  for (int r = 0; r < 8; ++r) {
    int m = mb + r;
    x[((size_t)b * CC_ + m) * PIX_ + p0 + n] = acc[r] + b_in[m];
  }
}

// ================= kernel 2: branch conv 16->16, KSxKS, implicit-GEMM WMMA ===========
// Full weight tensor (16*16*KS*KS, up to 50KB for KS=7) staged via one TDM per block.
template <int KS>
__global__ void branch_conv_wmma(const float* __restrict__ x,
                                 const float* __restrict__ w,
                                 const float* __restrict__ bias,
                                 float* __restrict__ out) {
  const int PAD  = KS / 2;
  const int KTOT = CC_ * KS * KS;
  const int KCH  = (KTOT + 31) / 32;
  __shared__ float wlds[CC_ * CC_ * KS * KS];
  stage_to_lds(wlds, w, CC_ * KTOT);

  int wave = (blockIdx.x * blockDim.x + threadIdx.x) >> 5;
  int lane = threadIdx.x & 31;
  const int wtiles = WW_ / 16;                 // 16
  int b   = wave / (HH_ * wtiles);
  int rem = wave % (HH_ * wtiles);
  int h   = rem / wtiles;
  int w0  = (rem % wtiles) * 16;
  int n   = lane & 15;

  v8f acc = {};
  for (int kc = 0; kc < KCH; ++kc) {
    v16h a, bf;
    #pragma unroll
    for (int i = 0; i < 16; ++i) {
      int ka = kc * 32 + a_kidx(lane, i);
      a[i] = (ka < KTOT) ? (_Float16)wlds[n * KTOT + ka] : (_Float16)0.f;
      int kb = kc * 32 + b_kidx(lane, i);
      _Float16 v = (_Float16)0.f;
      if (kb < KTOT) {
        int ci = kb / (KS * KS);
        int rr = kb % (KS * KS);
        int ky = rr / KS, kx = rr % KS;
        int ih = h + ky - PAD;
        int iw = w0 + n + kx - PAD;
        if (ih >= 0 && ih < HH_ && iw >= 0 && iw < WW_)
          v = (_Float16)x[((size_t)b * CC_ + ci) * PIX_ + ih * WW_ + iw];
      }
      bf[i] = v;
    }
    acc = __builtin_amdgcn_wmma_f32_16x16x32_f16(false, a, false, bf,
                                                 (short)0, acc, false, false);
  }
  int mb = (lane < 16) ? 0 : 8;
  #pragma unroll
  for (int r = 0; r < 8; ++r) {
    int m = mb + r;
    out[((size_t)b * CC_ + m) * PIX_ + h * WW_ + w0 + n] = acc[r] + bias[m];
  }
}

// ===== kernel 3: fused diff-kernels + expand(k3) + product + H-sort + SiLU + collapse(k4)
// One 256-thread block owns column (branch, b, group g, w); everything stays in LDS.
// softmax rows of k3 sum to 1 => grouped expand collapses to 8 neighbor FMAs;
// the 268MB/branch o1/o2/product tensors never touch HBM.
__global__ void branch_column_kernel(const float* __restrict__ c4,
                                     const float* __restrict__ k3s,
                                     const float* __restrict__ k4s,
                                     float* __restrict__ outs4) {
  __shared__ float colL[HH_], colC[HH_], colR[HH_];
  __shared__ float sb[HH_];
  __shared__ float k3row[EXP_ * 8];
  __shared__ float k4row[EXP_];

  const int br = blockIdx.y;
  const int d  = 2 * br + 1;                  // SHIFTS = {1,3,5,7}
  int idx = blockIdx.x;
  int w = idx % WW_;
  int g = (idx / WW_) % CC_;
  int b = idx / (WW_ * CC_);
  int t = threadIdx.x;                        // == h

  const float* cb = c4 + (((size_t)br * BB_ + b) * CC_ + g) * PIX_;
  colC[t] = cb[t * WW_ + w];
  colL[t] = (w - d >= 0)  ? cb[t * WW_ + (w - d)] : 0.f;
  colR[t] = (w + d < WW_) ? cb[t * WW_ + (w + d)] : 0.f;
  if (t < EXP_ * 8) k3row[t] = k3s[g * EXP_ * 8 + t];
  if (t < EXP_)     k4row[t] = k4s[g * EXP_ + t];
  __syncthreads();

  float nbr[8];
  bool um = (t - d) >= 0, dm = (t + d) < HH_;
  int hu = um ? (t - d) : 0;
  int hd = dm ? (t + d) : (HH_ - 1);
  nbr[0] = um ? colL[hu] : 0.f;   // (-d,-d)
  nbr[1] = um ? colC[hu] : 0.f;   // (-d, 0)
  nbr[2] = um ? colR[hu] : 0.f;   // (-d,+d)
  nbr[3] = colL[t];               // ( 0,-d)
  nbr[4] = dm ? colR[hd] : 0.f;   // (+d,+d)
  nbr[5] = dm ? colC[hd] : 0.f;   // (+d, 0)
  nbr[6] = dm ? colL[hd] : 0.f;   // (+d,-d)
  nbr[7] = colR[t];               // ( 0,+d)
  float cc = colC[t];

  float acc = 0.f;
  for (int e = 0; e < EXP_; ++e) {
    float o1 = 0.f, o2 = 0.f;
    #pragma unroll
    for (int j = 0; j < 8; ++j) {
      float kw = k3row[e * 8 + j];
      o1 += kw * (cc - nbr[j]);             // K1 order
      o2 += kw * (cc - nbr[(j + 4) & 7]);   // K2 = K1 rotated by 4
    }
    __syncthreads();
    sb[t] = o1 * o2;
    __syncthreads();
    // bitonic sort ascending over the H column (LDS-resident)
    for (int k = 2; k <= HH_; k <<= 1) {
      for (int j = k >> 1; j > 0; j >>= 1) {
        int ixj = t ^ j;
        if (ixj > t) {
          float av = sb[t], bv = sb[ixj];
          bool up = ((t & k) == 0);
          if (up ? (av > bv) : (av < bv)) { sb[t] = bv; sb[ixj] = av; }
        }
        __syncthreads();
      }
    }
    float v = sb[t];
    acc += k4row[e] * (v * sigmoidf_(v));   // SiLU then k4 collapse
  }
  outs4[(((size_t)br * BB_ + b) * CC_ + g) * PIX_ + t * WW_ + w] = acc;
}

// ================= kernel 4: max+mean over the 4 branches =================
__global__ void combine4_kernel(const float* __restrict__ outs4, float* __restrict__ tc) {
  size_t p = (size_t)blockIdx.x * 256 + threadIdx.x;
  const size_t N = (size_t)BB_ * CC_ * PIX_;
  float a = outs4[p], b = outs4[N + p], c = outs4[2 * N + p], d = outs4[3 * N + p];
  float mx = fmaxf(fmaxf(a, b), fmaxf(c, d));
  tc[p] = mx + (a + b + c + d) * 0.25f;
}

// ================= kernel 5: 3x3 conv 16->16 (WMMA) + BN(eval) + SiLU ================
__global__ void base_conv_bn_silu_wmma(const float* __restrict__ tin,
                                       const float* __restrict__ w,
                                       const float* __restrict__ gamma,
                                       const float* __restrict__ beta,
                                       const float* __restrict__ mean,
                                       const float* __restrict__ var,
                                       float* __restrict__ hb) {
  const int KS = 3, PAD = 1;
  const int KTOT = CC_ * KS * KS;             // 144
  const int KCH  = (KTOT + 31) / 32;          // 5
  __shared__ float wlds[CC_ * CC_ * 9];
  stage_to_lds(wlds, w, CC_ * KTOT);

  int wave = (blockIdx.x * blockDim.x + threadIdx.x) >> 5;
  int lane = threadIdx.x & 31;
  const int wtiles = WW_ / 16;
  int b   = wave / (HH_ * wtiles);
  int rem = wave % (HH_ * wtiles);
  int h   = rem / wtiles;
  int w0  = (rem % wtiles) * 16;
  int n   = lane & 15;

  v8f acc = {};
  for (int kc = 0; kc < KCH; ++kc) {
    v16h a, bf;
    #pragma unroll
    for (int i = 0; i < 16; ++i) {
      int ka = kc * 32 + a_kidx(lane, i);
      a[i] = (ka < KTOT) ? (_Float16)wlds[n * KTOT + ka] : (_Float16)0.f;
      int kb = kc * 32 + b_kidx(lane, i);
      _Float16 v = (_Float16)0.f;
      if (kb < KTOT) {
        int ci = kb / 9, rr = kb % 9;
        int ky = rr / 3, kx = rr % 3;
        int ih = h + ky - PAD;
        int iw = w0 + n + kx - PAD;
        if (ih >= 0 && ih < HH_ && iw >= 0 && iw < WW_)
          v = (_Float16)tin[((size_t)b * CC_ + ci) * PIX_ + ih * WW_ + iw];
      }
      bf[i] = v;
    }
    acc = __builtin_amdgcn_wmma_f32_16x16x32_f16(false, a, false, bf,
                                                 (short)0, acc, false, false);
  }
  int mb = (lane < 16) ? 0 : 8;
  #pragma unroll
  for (int r = 0; r < 8; ++r) {
    int m = mb + r;
    float v = acc[r];
    v = (v - mean[m]) * __frsqrt_rn(var[m] + BN_EPS_) * gamma[m] + beta[m];
    v = v * sigmoidf_(v);
    hb[((size_t)b * CC_ + m) * PIX_ + h * WW_ + w0 + n] = v;
  }
}

// ================= kernel 6: 1x1 conv 16->1 + sigmoid =================
__global__ void out_mask_kernel(const float* __restrict__ hb,
                                const float* __restrict__ w_out,
                                const float* __restrict__ b_out,
                                float* __restrict__ mask) {
  int p = blockIdx.x * 256 + threadIdx.x;     // B*PIX
  int b = p >> 16, hw = p & (PIX_ - 1);
  float s = b_out[0];
  #pragma unroll
  for (int c = 0; c < CC_; ++c)
    s += w_out[c] * hb[((size_t)b * CC_ + c) * PIX_ + hw];
  mask[p] = sigmoidf_(s);
}

// ================= kernel 7: final elementwise blend =================
__global__ void final_kernel(const float* __restrict__ cen,
                             const float* __restrict__ mas,
                             const float* __restrict__ mask,
                             const float* __restrict__ s1,
                             float* __restrict__ out) {
  size_t p = (size_t)blockIdx.x * 256 + threadIdx.x;   // B*64*PIX
  int hw = (int)(p & (PIX_ - 1));
  int b  = (int)(p >> 22);                              // p / (64*65536)
  float mk = mask[(b << 16) + hw];
  float mm = sigmoidf_(mas[(b << 16) + hw]);
  out[p] = cen[p] * (mk * mm * s1[0] + mm * s1[1] + s1[2] * mk + s1[3]);
}

// ====================================================================
extern "C" void kernel_launch(void* const* d_in, const int* in_sizes, int n_in,
                              void* d_out, int out_size, void* d_ws, size_t ws_size,
                              hipStream_t stream) {
  const float* cen      = (const float*)d_in[0];
  const float* mas      = (const float*)d_in[1];
  const float* w_in     = (const float*)d_in[2];
  const float* b_in     = (const float*)d_in[3];
  const float* w_c0     = (const float*)d_in[4];
  const float* b_c0     = (const float*)d_in[5];
  const float* w_c1     = (const float*)d_in[6];
  const float* b_c1     = (const float*)d_in[7];
  const float* w_c2     = (const float*)d_in[8];
  const float* b_c2     = (const float*)d_in[9];
  const float* w_c3     = (const float*)d_in[10];
  const float* b_c3     = (const float*)d_in[11];
  const float* scales1  = (const float*)d_in[12];
  const float* scales2  = (const float*)d_in[13];
  const float* scales3  = (const float*)d_in[14];
  const float* w_base   = (const float*)d_in[15];
  const float* bn_gamma = (const float*)d_in[16];
  const float* bn_beta  = (const float*)d_in[17];
  const float* bn_mean  = (const float*)d_in[18];
  const float* bn_var   = (const float*)d_in[19];
  const float* w_out    = (const float*)d_in[20];
  const float* b_out    = (const float*)d_in[21];
  float* outp = (float*)d_out;

  // -------- workspace carve (floats) --------
  const size_t NF = (size_t)BB_ * CC_ * PIX_;          // 4.19M per [B,C,H,W]
  float* ws   = (float*)d_ws;
  float* x    = ws;                 ws += NF;          // in_conv output
  float* c4   = ws;                 ws += 4 * NF;      // per-branch conv outputs
  float* o4   = ws;                 ws += 4 * NF;      // per-branch module outputs
  float* tc   = ws;                 ws += NF;          // max+mean combined
  float* hb   = ws;                 ws += NF;          // base conv output
  float* mk   = ws;                 ws += (size_t)BB_ * PIX_;
  float* k3s  = ws;                 ws += 256 * 8;
  float* k4s  = ws;                 ws += CC_ * EXP_;
  float* s1b  = ws;                 ws += 4;
  (void)ws_size; (void)in_sizes; (void)n_in; (void)out_size;

  // -------- pipeline --------
  softmax_small_kernel<<<1, 256, 0, stream>>>(scales1, scales2, scales3, s1b, k3s, k4s);

  in_conv_wmma<<<2048, 256, 0, stream>>>(cen, w_in, b_in, x);   // 16384 waves

  branch_conv_wmma<1><<<2048, 256, 0, stream>>>(x, w_c0, b_c0, c4 + 0 * NF);
  branch_conv_wmma<3><<<2048, 256, 0, stream>>>(x, w_c1, b_c1, c4 + 1 * NF);
  branch_conv_wmma<5><<<2048, 256, 0, stream>>>(x, w_c2, b_c2, c4 + 2 * NF);
  branch_conv_wmma<7><<<2048, 256, 0, stream>>>(x, w_c3, b_c3, c4 + 3 * NF);

  dim3 gcol(BB_ * CC_ * WW_, 4);                                 // 16384 cols x 4 branches
  branch_column_kernel<<<gcol, 256, 0, stream>>>(c4, k3s, k4s, o4);

  combine4_kernel<<<(unsigned)(NF / 256), 256, 0, stream>>>(o4, tc);

  base_conv_bn_silu_wmma<<<2048, 256, 0, stream>>>(tc, w_base, bn_gamma, bn_beta,
                                                   bn_mean, bn_var, hb);

  out_mask_kernel<<<(BB_ * PIX_) / 256, 256, 0, stream>>>(hb, w_out, b_out, mk);

  final_kernel<<<(unsigned)(((size_t)BB_ * CIN_ * PIX_) / 256), 256, 0, stream>>>(
      cen, mas, mk, s1b, outp);
}
